// MambaBlock_28527172780514
// MI455X (gfx1250) — matrix operand order repaired and checked by reading, not compiled, not run on hardware
//
#include <hip/hip_runtime.h>

typedef __attribute__((ext_vector_type(16))) __bf16 v16bf;
typedef __attribute__((ext_vector_type(8)))  __bf16 v8bf;
typedef __attribute__((ext_vector_type(8)))  float  v8f;

#define BQ    2
#define LQ    1024
#define DM    1024
#define DI    2048
#define DSN   16
#define DTR   64
#define KC    4
#define NT    (BQ * LQ)          /* 2048 tokens */
#define XCOLS (DTR + 2 * DSN)    /* 96 */

// ---------------------------------------------------------------------------
// fp32 [Kd x N] -> bf16 transposed [N x Kd]  (coalesced writes)
// ---------------------------------------------------------------------------
__global__ __launch_bounds__(256) void k_convt(const float* __restrict__ in,
                                               __bf16* __restrict__ out,
                                               int Kd, int N) {
    int idx = blockIdx.x * 256 + threadIdx.x;
    if (idx >= Kd * N) return;
    int n = idx / Kd, k = idx % Kd;
    out[idx] = (__bf16)in[(size_t)k * N + n];
}

// ---------------------------------------------------------------------------
// RMSNorm per token (block = 256 thr, one token per block), emits bf16
// ---------------------------------------------------------------------------
__global__ __launch_bounds__(256) void k_rmsnorm(const float* __restrict__ x0,
                                                 const float* __restrict__ w,
                                                 __bf16* __restrict__ h) {
    int t = blockIdx.x;
    const float* xp = x0 + (size_t)t * DM;
    float ss = 0.f;
    for (int i = threadIdx.x; i < DM; i += 256) { float v = xp[i]; ss += v * v; }
    ss += __shfl_xor(ss, 16, 32);
    ss += __shfl_xor(ss,  8, 32);
    ss += __shfl_xor(ss,  4, 32);
    ss += __shfl_xor(ss,  2, 32);
    ss += __shfl_xor(ss,  1, 32);
    __shared__ float sred[8];
    if ((threadIdx.x & 31) == 0) sred[threadIdx.x >> 5] = ss;
    __syncthreads();
    float tot = (threadIdx.x < 8) ? sred[threadIdx.x] : 0.f;
    tot += __shfl_xor(tot, 4, 32);
    tot += __shfl_xor(tot, 2, 32);
    tot += __shfl_xor(tot, 1, 32);
    __shared__ float s_scale;
    if (threadIdx.x == 0) s_scale = rsqrtf(tot / (float)DM + 1e-5f);
    __syncthreads();
    float sc = s_scale;
    for (int i = threadIdx.x; i < DM; i += 256)
        h[(size_t)t * DM + i] = (__bf16)(xp[i] * sc * w[i]);
}

// ---------------------------------------------------------------------------
// WMMA bf16 GEMM: C[MxN] = A[MxK] @ B[KxN], B pre-transposed as Bt[NxK].
// Each wave owns an MT x NTW grid of 16x16 C tiles (register blocking:
// MT+NTW fragment loads feed MT*NTW WMMAs per K-step). 4 waves/block along N.
// Fragment layouts follow the CDNA5 ISA 16-bit A(16x32)/B(32x16) lane maps.
// ---------------------------------------------------------------------------
template <int MT, int NTW>
__global__ __launch_bounds__(128) void k_gemm(const __bf16* __restrict__ A,
                                              const __bf16* __restrict__ Bt,
                                              float* __restrict__ C,
                                              int M, int N, int Kd) {
    const int lane = threadIdx.x & 31;
    const int half = lane >> 4;
    const int r15  = lane & 15;
    const int tn0  = (blockIdx.x * blockDim.y + threadIdx.y) * NTW;  // N tile base
    const int tm0  = blockIdx.y * MT;                                // M tile base
    if (tn0 * 16 >= N) return;   // wave-uniform -> EXEC stays all-ones

    const __bf16* ap[MT];
    const __bf16* bp[NTW];
#pragma unroll
    for (int i = 0; i < MT; ++i)
        // A: lanes 0-15: row=r15, K={0..7,16..23}; lanes 16-31: K={8..15,24..31}
        ap[i] = A + (size_t)((tm0 + i) * 16 + r15) * Kd + half * 8;
#pragma unroll
    for (int j = 0; j < NTW; ++j)
        // Bt: lanes 0-15: col=r15, K={0..7,16..23}; lanes 16-31: K={8..15,24..31}
        bp[j] = Bt + (size_t)((tn0 + j) * 16 + r15) * Kd + half * 8;

    v8f acc[MT][NTW] = {};
    for (int kk = 0; kk < Kd; kk += 32) {
        v16bf a[MT], b[NTW];
#pragma unroll
        for (int i = 0; i < MT; ++i) {
            v8bf a0 = *(const v8bf*)(ap[i] + kk);
            v8bf a1 = *(const v8bf*)(ap[i] + kk + 16);
#pragma unroll
            for (int e = 0; e < 8; ++e) { a[i][e] = a0[e]; a[i][e + 8] = a1[e]; }
        }
#pragma unroll
        for (int j = 0; j < NTW; ++j) {
            v8bf b0 = *(const v8bf*)(bp[j] + kk);
            v8bf b1 = *(const v8bf*)(bp[j] + kk + 16);
#pragma unroll
            for (int e = 0; e < 8; ++e) { b[j][e] = b0[e]; b[j][e + 8] = b1[e]; }
        }
#pragma unroll
        for (int i = 0; i < MT; ++i)
#pragma unroll
            for (int j = 0; j < NTW; ++j)
                acc[i][j] = __builtin_amdgcn_wmma_f32_16x16x32_bf16(
                    false, a[i], false, b[j], (short)0, acc[i][j], false, false);
    }
    // C/D layout: VGPR v -> row = v + 8*half, col = r15
#pragma unroll
    for (int i = 0; i < MT; ++i)
#pragma unroll
        for (int j = 0; j < NTW; ++j) {
            float* cp = C + (size_t)((tm0 + i) * 16 + half * 8) * N + (tn0 + j) * 16 + r15;
#pragma unroll
            for (int v = 0; v < 8; ++v) cp[(size_t)v * N] = acc[i][j][v];
        }
}

// ---------------------------------------------------------------------------
// Causal depthwise conv (K=4) + bias + SiLU; emits fp32 and bf16 copies
// ---------------------------------------------------------------------------
__global__ __launch_bounds__(256) void k_conv_silu(const float* __restrict__ xz,
                                                   const float* __restrict__ cw,
                                                   const float* __restrict__ cb,
                                                   float* __restrict__ xact,
                                                   __bf16* __restrict__ xact_bf) {
    int idx = blockIdx.x * 256 + threadIdx.x;     // over B*L*DI
    if (idx >= NT * DI) return;
    int d = idx % DI;
    int t = idx / DI;
    int l = t % LQ, b = t / LQ;
    float acc = cb[d];
#pragma unroll
    for (int j = 0; j < KC; ++j) {
        int ls = l - (KC - 1) + j;
        if (ls >= 0)
            acc += xz[((size_t)(b * LQ + ls)) * (2 * DI) + d] * cw[d * KC + j];
    }
    float s = acc / (1.f + __expf(-acc));
    xact[idx]    = s;
    xact_bf[idx] = (__bf16)s;
}

// dlt columns [0:64) of xdbl -> bf16 A-matrix for dt-proj GEMM
__global__ __launch_bounds__(256) void k_extract_dlt(const float* __restrict__ xdbl,
                                                     __bf16* __restrict__ dlt) {
    int idx = blockIdx.x * 256 + threadIdx.x;     // over NT*DTR
    if (idx >= NT * DTR) return;
    int t = idx / DTR, r = idx % DTR;
    dlt[idx] = (__bf16)xdbl[(size_t)t * XCOLS + r];
}

// delta = softplus(raw + b_dt), in place
__global__ __launch_bounds__(256) void k_softplus(float* __restrict__ delta,
                                                  const float* __restrict__ b_dt) {
    int idx = blockIdx.x * 256 + threadIdx.x;     // over NT*DI
    if (idx >= NT * DI) return;
    float v  = delta[idx] + b_dt[idx % DI];
    delta[idx] = (v > 20.f) ? v : log1pf(__expf(v));
}

// ---------------------------------------------------------------------------
// Selective scan: lane = 16*half + n; one wave handles 2 channels of a batch.
// Per step: 1 exp + 2 fma per lane, then 16-lane shfl_xor reduction for y.
// ---------------------------------------------------------------------------
__global__ __launch_bounds__(256) void k_scan(const float* __restrict__ delta,
                                              const float* __restrict__ xact,
                                              const float* __restrict__ xdbl,
                                              const float* __restrict__ Amat,
                                              float* __restrict__ ys) {
    const int lane = threadIdx.x & 31;
    const int wav  = threadIdx.x >> 5;
    const int p    = blockIdx.x * 8 + wav;        // 0 .. B*DI/2-1
    const int half = lane >> 4;
    const int n    = lane & 15;
    const int b    = p / (DI / 2);
    const int d    = (p % (DI / 2)) * 2 + half;
    const float a  = Amat[d * DSN + n];
    const float* dp = delta + (size_t)b * LQ * DI + d;
    const float* xp = xact  + (size_t)b * LQ * DI + d;
    const float* bc = xdbl  + (size_t)b * LQ * XCOLS;
    float*       yp = ys    + (size_t)b * LQ * DI + d;
    float s = 0.f;
    for (int l = 0; l < LQ; ++l) {
        float dv = dp[(size_t)l * DI];
        float xv = xp[(size_t)l * DI];
        float Bv = bc[l * XCOLS + DTR + n];
        float Cv = bc[l * XCOLS + DTR + DSN + n];
        float dA = __expf(dv * a);
        s = fmaf(dA, s, dv * xv * Bv);
        float prod = s * Cv;
        prod += __shfl_xor(prod, 8, 32);
        prod += __shfl_xor(prod, 4, 32);
        prod += __shfl_xor(prod, 2, 32);
        prod += __shfl_xor(prod, 1, 32);
        if (n == 0) yp[(size_t)l * DI] = prod;
    }
}

// y = (ys + x*D) * silu(res)   -> bf16 A-matrix for out-proj GEMM
__global__ __launch_bounds__(256) void k_gate(const float* __restrict__ ys,
                                              const float* __restrict__ xact,
                                              const float* __restrict__ Dp,
                                              const float* __restrict__ xz,
                                              __bf16* __restrict__ yg) {
    int idx = blockIdx.x * 256 + threadIdx.x;     // over NT*DI
    if (idx >= NT * DI) return;
    int d = idx % DI;
    int t = idx / DI;
    float r   = xz[(size_t)t * (2 * DI) + DI + d];
    float sil = r / (1.f + __expf(-r));
    yg[idx] = (__bf16)((ys[idx] + xact[idx] * Dp[d]) * sil);
}

// out = x0 + mm
__global__ __launch_bounds__(256) void k_addres(const float* __restrict__ x0,
                                                const float* __restrict__ mm,
                                                float* __restrict__ out) {
    int idx = blockIdx.x * 256 + threadIdx.x;     // over NT*DM
    if (idx >= NT * DM) return;
    out[idx] = x0[idx] + mm[idx];
}

// ---------------------------------------------------------------------------
extern "C" void kernel_launch(void* const* d_in, const int* in_sizes, int n_in,
                              void* d_out, int out_size, void* d_ws, size_t ws_size,
                              hipStream_t stream) {
    (void)in_sizes; (void)n_in; (void)out_size; (void)ws_size;
    const float* x0    = (const float*)d_in[0];
    const float* normw = (const float*)d_in[1];
    const float* W_in  = (const float*)d_in[2];
    const float* convw = (const float*)d_in[3];
    const float* convb = (const float*)d_in[4];
    const float* W_x   = (const float*)d_in[5];
    const float* W_dt  = (const float*)d_in[6];
    const float* b_dt  = (const float*)d_in[7];
    const float* Amat  = (const float*)d_in[8];
    const float* Dvec  = (const float*)d_in[9];
    const float* W_out = (const float*)d_in[10];
    float* out = (float*)d_out;

    char* ws = (char*)d_ws;
    size_t off = 0;
    auto alloc = [&](size_t bytes) -> void* {
        void* p = ws + off;
        off = (off + bytes + 255) & ~(size_t)255;
        return p;
    };
    __bf16* h_bf    = (__bf16*)alloc((size_t)NT * DM * 2);
    __bf16* Win_t   = (__bf16*)alloc((size_t)(2 * DI) * DM * 2);
    __bf16* Wx_t    = (__bf16*)alloc((size_t)XCOLS * DI * 2);
    __bf16* Wdt_t   = (__bf16*)alloc((size_t)DI * DTR * 2);
    __bf16* Wout_t  = (__bf16*)alloc((size_t)DM * DI * 2);
    float*  xz      = (float*) alloc((size_t)NT * 2 * DI * 4);
    float*  xact    = (float*) alloc((size_t)NT * DI * 4);
    __bf16* xact_bf = (__bf16*)alloc((size_t)NT * DI * 2);
    float*  xdbl    = (float*) alloc((size_t)NT * XCOLS * 4);
    __bf16* dlt_bf  = (__bf16*)alloc((size_t)NT * DTR * 2);
    float*  deltab  = (float*) alloc((size_t)NT * DI * 4);
    float*  yscan   = (float*) alloc((size_t)NT * DI * 4);
    __bf16* yg_bf   = (__bf16*)alloc((size_t)NT * DI * 2);
    float*  mmout   = (float*) alloc((size_t)NT * DM * 4);

    const dim3 gblk(32, 4);   // 4 waves per block along N

    // weight conversions (fp32 -> bf16, transposed to [N x K], coalesced writes)
    k_convt<<<(DM * 2 * DI + 255) / 256, 256, 0, stream>>>(W_in,  Win_t,  DM,  2 * DI);
    k_convt<<<(DI * XCOLS  + 255) / 256, 256, 0, stream>>>(W_x,   Wx_t,   DI,  XCOLS);
    k_convt<<<(DTR * DI    + 255) / 256, 256, 0, stream>>>(W_dt,  Wdt_t,  DTR, DI);
    k_convt<<<(DI * DM     + 255) / 256, 256, 0, stream>>>(W_out, Wout_t, DI,  DM);

    // 1. RMSNorm -> bf16
    k_rmsnorm<<<NT, 256, 0, stream>>>(x0, normw, h_bf);
    // 2. in-proj GEMM: [2048x1024] @ [1024x4096]; wave tile 32x64, block tile 32x256
    k_gemm<2, 4><<<dim3((2 * DI) / 256, NT / 32), gblk, 0, stream>>>(h_bf, Win_t, xz, NT, 2 * DI, DM);
    // 3. causal depthwise conv + SiLU
    k_conv_silu<<<(NT * DI + 255) / 256, 256, 0, stream>>>(xz, convw, convb, xact, xact_bf);
    // 4. x-proj GEMM: [2048x2048] @ [2048x96] (6 N-tiles, 1x1 tiles/wave, guard)
    k_gemm<1, 1><<<dim3(2, NT / 16), gblk, 0, stream>>>(xact_bf, Wx_t, xdbl, NT, XCOLS, DI);
    // 5. dt-proj GEMM: [2048x64] @ [64x2048], then softplus(+b_dt) in place
    k_extract_dlt<<<(NT * DTR + 255) / 256, 256, 0, stream>>>(xdbl, dlt_bf);
    k_gemm<2, 4><<<dim3(DI / 256, NT / 32), gblk, 0, stream>>>(dlt_bf, Wdt_t, deltab, NT, DI, DTR);
    k_softplus<<<(NT * DI + 255) / 256, 256, 0, stream>>>(deltab, b_dt);
    // 6. selective scan (2048 waves, 2 channels/wave, state-per-lane)
    k_scan<<<(BQ * DI / 2) / 8, 256, 0, stream>>>(deltab, xact, xdbl, Amat, yscan);
    // 7. gating -> bf16
    k_gate<<<(NT * DI + 255) / 256, 256, 0, stream>>>(yscan, xact, Dvec, xz, yg_bf);
    // 8. out-proj GEMM: [2048x2048] @ [2048x1024]
    k_gemm<2, 4><<<dim3(DM / 256, NT / 32), gblk, 0, stream>>>(yg_bf, Wout_t, mmout, NT, DM, DI);
    // 9. residual
    k_addres<<<(NT * DM + 255) / 256, 256, 0, stream>>>(x0, mmout, out);
}
